// CustomLSTM_63058709840651
// MI455X (gfx1250) — matrix operand order repaired and checked by reading, not compiled, not run on hardware
//
#include <hip/hip_runtime.h>
#include <hip/hip_bf16.h>
#include <math.h>

// Problem sizes (fixed by the reference)
#define BQ 256
#define TQ 1024
#define IQ 64
#define HQ 256
#define OQ 8

typedef __attribute__((ext_vector_type(16))) __bf16 v16bf;
typedef __attribute__((ext_vector_type(8)))  float  v8f;
typedef __attribute__((ext_vector_type(4)))  int    v4i;
// Global-address-space 16B vector for explicit global_load_b128 generation.
typedef __attribute__((address_space(1))) const v4i gc_v4i;

// Fast transcendentals: gfx1250 has a hardware V_TANH_F32.
#if __has_builtin(__builtin_amdgcn_tanhf)
  #define FAST_TANH(x) __builtin_amdgcn_tanhf(x)
#elif __has_builtin(__builtin_amdgcn_tanh_f32)
  #define FAST_TANH(x) __builtin_amdgcn_tanh_f32(x)
#else
  #define FAST_TANH(x) tanhf(x)
#endif
// sigmoid(x) = 0.5*tanh(x/2) + 0.5  (mul + v_tanh + fma)
__device__ __forceinline__ float fast_sigmoid(float x) {
    return fmaf(0.5f, FAST_TANH(0.5f * x), 0.5f);
}

__device__ __forceinline__ unsigned short f2bf_rne(float f) {
    unsigned u = __float_as_uint(f);
    unsigned r = u + 0x7FFFu + ((u >> 16) & 1u);
    return (unsigned short)(r >> 16);
}

// ---------------------------------------------------------------------------
// Pack Wg = [W_f|W_i|W_c|W_o] (each [320,256] f32 row-major, rows 0..255 = h
// part, 256..319 = x part) into bf16 B-matrix fragments.
// Fragment layout (16-bit B, 32x16): lane L holds column N = L%16; bf16 j of
// the 16-vector holds K_local = (j%8) + 8*(L>>4) + 16*(j>>3). Storage:
//   wpk[((nt*10 + kt)*32 + lane)*16 + j], nt = gate*16 + coltile, kt = K/32.
// ---------------------------------------------------------------------------
__global__ void pack_w_kernel(const float* __restrict__ Wf, const float* __restrict__ Wi,
                              const float* __restrict__ Wc, const float* __restrict__ Wo,
                              unsigned short* __restrict__ wpk) {
    int idx = blockIdx.x * blockDim.x + threadIdx.x;
    if (idx >= 64 * 10 * 512) return;
    int j     = idx & 15;
    int lane  = (idx >> 4) & 31;
    int rest  = idx >> 9;
    int kt    = rest % 10;
    int nt    = rest / 10;
    int g     = nt >> 4;
    int col   = ((nt & 15) << 4) + (lane & 15);
    int k     = kt * 32 + (j & 7) + 8 * (lane >> 4) + 16 * (j >> 3);   // 0..319
    const float* W = (g == 0) ? Wf : (g == 1) ? Wi : (g == 2) ? Wc : Wo;
    wpk[idx] = f2bf_rne(W[k * HQ + col]);
}

// ---------------------------------------------------------------------------
// Pack x [256,1024,64] f32 into bf16 A-matrix fragments per (batch-tile, t).
// Fragment layout (16-bit A, 16x32): lane L holds row M = L%16; bf16 j holds
// K_local = (j%8) + 8*(L>>4) + 16*(j>>3). Storage:
//   xpk[(((btile*T + t)*2 + kt)*32 + lane)*16 + j]
// ---------------------------------------------------------------------------
__global__ void pack_x_kernel(const float* __restrict__ x, unsigned short* __restrict__ xpk) {
    int idx = blockIdx.x * blockDim.x + threadIdx.x;   // 16*1024*2*512 = 2^24
    int j     = idx & 15;
    int lane  = (idx >> 4) & 31;
    int kt    = (idx >> 9) & 1;
    int t     = (idx >> 10) & 1023;
    int btile = idx >> 20;
    int m     = lane & 15;
    int K     = (j & 7) + 8 * (lane >> 4) + 16 * (j >> 3);
    int xcol  = kt * 32 + K;                           // 0..63
    xpk[idx] = f2bf_rne(x[((size_t)(btile * 16 + m) * TQ + t) * IQ + xcol]);
}

// ---------------------------------------------------------------------------
// Persistent LSTM kernel: 16 blocks (one per 16-row batch tile) x 512 threads
// (16 waves; wave w owns hidden columns [16w,16w+16) for all 4 gates).
// c lives in WMMA accumulator registers; h lives in double-buffered bf16 LDS.
// One workgroup barrier per timestep. The x-contribution for step t+1 is
// computed before the barrier of step t (it does not depend on h), hiding its
// loads + 8 WMMAs behind the barrier wait. Bias enters as the C operand of the
// first x-WMMA; the x-partial enters as the C operand of the first h-WMMA, so
// no accumulator copies are needed (VOP3P WMMA has separate VDST and SRC2).
// Weight/x bases are laundered u64s rebuilt as addrspace(1) pointers each
// iteration: loads stay in-loop (no 320-VGPR spill) AND lower to
// global_load_b128 (LOADcnt only), not flat_load (LOADcnt+DScnt, which would
// couple the weight stream into the ds-store/barrier wait).
// ---------------------------------------------------------------------------
__global__ __launch_bounds__(512)
void lstm_persistent_kernel(const unsigned short* __restrict__ xpk,
                            const unsigned short* __restrict__ wpk,
                            const float* __restrict__ bfv, const float* __restrict__ biv,
                            const float* __restrict__ bcv, const float* __restrict__ bov,
                            const float* __restrict__ Who, const float* __restrict__ bho,
                            float* __restrict__ out) {
    __shared__ unsigned short h_lds[2][16][HQ];   // 16 KiB, double-buffered bf16 h
    __shared__ float          h_out[16][HQ];      // 16 KiB, final f32 h_T

    const int tid   = threadIdx.x;
    const int lane  = tid & 31;
    const int w     = tid >> 5;          // wave id 0..15 -> hidden column tile
    const int btile = blockIdx.x;        // 0..15

    // h_0 = 0
    for (int i = tid; i < 16 * HQ; i += 512) ((unsigned short*)h_lds[0])[i] = 0;
    __syncthreads();

    const int m    = lane & 15;          // A-matrix row / C-matrix lane group
    const int hi   = lane >> 4;
    const int ncol = (w << 4) + m;       // this lane's hidden column (C/D: N = lane%16)

    // Persistent bias accumulators (C/D layout: same value in all 8 VGPRs).
    v8f biasacc[4];
    {
        const float bv[4] = { bfv[ncol], biv[ncol], bcv[ncol], bov[ncol] };
        #pragma unroll
        for (int g = 0; g < 4; ++g)
            for (int r = 0; r < 8; ++r) biasacc[g][r] = bv[g];
    }

    v8f c_state; for (int r = 0; r < 8; ++r) c_state[r] = 0.f;
    v8f h_last;  for (int r = 0; r < 8; ++r) h_last[r]  = 0.f;

    // Laundered base addresses (SGPR u64) -> rebuilt as global pointers.
    unsigned long long wbase = (unsigned long long)wpk;
    unsigned long long xbase = (unsigned long long)(xpk + (size_t)btile * TQ * 2 * 512);
    const unsigned laneoff = (unsigned)lane * 32u;   // per-lane byte offset

    // Helper lambdas -------------------------------------------------------
    auto loadB = [&](unsigned long long base, int nt, int kt) {
        v16bf b;
        unsigned long long addr = base + (unsigned)((nt * 10 + kt) * 1024) + laneoff;
        ((v4i*)&b)[0] = *(gc_v4i*)(addr);
        ((v4i*)&b)[1] = *(gc_v4i*)(addr + 16);
        return b;
    };
    auto loadAx = [&](unsigned long long base, int t, int kt) {
        v16bf a;
        unsigned long long addr = base + (unsigned)((t * 2 + kt) * 1024) + laneoff;
        ((v4i*)&a)[0] = *(gc_v4i*)(addr);
        ((v4i*)&a)[1] = *(gc_v4i*)(addr + 16);
        return a;
    };

    v8f xacc[4];   // bias + x-part partial accumulators for the *next* step

    // ---- prologue: x-partials for t = 0 (C operand = biasacc, no copies) ----
    {
        v16bf a0 = loadAx(xbase, 0, 0);
        v16bf a1 = loadAx(xbase, 0, 1);
        #pragma unroll
        for (int g = 0; g < 4; ++g) {
            const int nt = (g << 4) + w;
            xacc[g] = __builtin_amdgcn_wmma_f32_16x16x32_bf16(
                false, a0, false, loadB(wbase, nt, 8), (short)0, biasacc[g], false, false);
            xacc[g] = __builtin_amdgcn_wmma_f32_16x16x32_bf16(
                false, a1, false, loadB(wbase, nt, 9), (short)0, xacc[g], false, false);
        }
    }

    for (int t = 0; t < TQ; ++t) {
        const int rb = t & 1, wb = rb ^ 1;

        // Launder bases: keep fragment loads inside the loop, no hoist/spill.
        asm volatile("" : "+s"(wbase));
        asm volatile("" : "+s"(xbase));

        // ---- h-part of the GEMM: K tiles 0..7 (A from LDS, B from L2). ----
        // First K-tile consumes xacc as C, so acc needs no init copy.
        v8f acc[4];
        #pragma unroll
        for (int kt = 0; kt < 8; ++kt) {
            v16bf a;
            const int k0 = kt * 32 + hi * 8;
            ((int4*)&a)[0] = *(const int4*)&h_lds[rb][m][k0];
            ((int4*)&a)[1] = *(const int4*)&h_lds[rb][m][k0 + 16];
            #pragma unroll
            for (int g = 0; g < 4; ++g) {
                const int nt = (g << 4) + w;
                acc[g] = __builtin_amdgcn_wmma_f32_16x16x32_bf16(
                    false, a, false, loadB(wbase, nt, kt), (short)0,
                    (kt == 0) ? xacc[g] : acc[g], false, false);
            }
        }

        // ---- gate nonlinearities + state update ----
        // C/D layout: VGPR r -> M = r + 8*(lane>>4), N = lane%16
        #pragma unroll
        for (int r = 0; r < 8; ++r) {
            float fg = fast_sigmoid(acc[0][r]);
            float ig = fast_sigmoid(acc[1][r]);
            float cg = FAST_TANH(acc[2][r]);
            float og = fast_sigmoid(acc[3][r]);
            float cn = fmaf(fg, c_state[r], ig * cg);
            c_state[r] = cn;
            float hn = og * FAST_TANH(cn);
            h_last[r] = hn;
            h_lds[wb][r + 8 * hi][ncol] = f2bf_rne(hn);
        }

        // ---- x-partials for step t+1 (independent of h; overlaps barrier) ----
        if (t + 1 < TQ) {
            v16bf a0 = loadAx(xbase, t + 1, 0);
            v16bf a1 = loadAx(xbase, t + 1, 1);
            #pragma unroll
            for (int g = 0; g < 4; ++g) {
                const int nt = (g << 4) + w;
                xacc[g] = __builtin_amdgcn_wmma_f32_16x16x32_bf16(
                    false, a0, false, loadB(wbase, nt, 8), (short)0, biasacc[g], false, false);
                xacc[g] = __builtin_amdgcn_wmma_f32_16x16x32_bf16(
                    false, a1, false, loadB(wbase, nt, 9), (short)0, xacc[g], false, false);
            }
        }

        __syncthreads();   // writes to wb visible before next step reads it
    }

    // ---- final projection: out[16,8] = h_T @ W_ho + b_ho ----
    #pragma unroll
    for (int r = 0; r < 8; ++r) h_out[r + 8 * hi][ncol] = h_last[r];
    __syncthreads();
    if (tid < 16 * OQ) {
        const int mo = tid >> 3, o = tid & 7;
        float s = bho[o];
        for (int k = 0; k < HQ; ++k) s += h_out[mo][k] * Who[k * OQ + o];
        out[(btile * 16 + mo) * OQ + o] = s;
    }
}

// ---------------------------------------------------------------------------
extern "C" void kernel_launch(void* const* d_in, const int* in_sizes, int n_in,
                              void* d_out, int out_size, void* d_ws, size_t ws_size,
                              hipStream_t stream) {
    const float* x   = (const float*)d_in[0];
    const float* Wf  = (const float*)d_in[1];
    const float* bf_ = (const float*)d_in[2];
    const float* Wi  = (const float*)d_in[3];
    const float* bi_ = (const float*)d_in[4];
    const float* Wc  = (const float*)d_in[5];
    const float* bc_ = (const float*)d_in[6];
    const float* Wo  = (const float*)d_in[7];
    const float* bo_ = (const float*)d_in[8];
    const float* Who = (const float*)d_in[9];
    const float* bho = (const float*)d_in[10];

    const size_t xpk_elems = (size_t)16 * TQ * 2 * 512;      // 16,777,216 bf16 = 32 MiB
    unsigned short* xpk = (unsigned short*)d_ws;
    unsigned short* wpk = (unsigned short*)((char*)d_ws + xpk_elems * sizeof(unsigned short));

    pack_x_kernel<<<(int)(xpk_elems / 256), 256, 0, stream>>>(x, xpk);
    pack_w_kernel<<<(64 * 10 * 512) / 256, 256, 0, stream>>>(Wf, Wi, Wc, Wo, wpk);
    lstm_persistent_kernel<<<16, 512, 0, stream>>>(xpk, wpk, bf_, bi_, bc_, bo_,
                                                   Who, bho, (float*)d_out);
}
